// CSWinBlock_62371515073110
// MI455X (gfx1250) — compile-verified, matrix operand order checked by reading, not run on hardware
//
#include <hip/hip_runtime.h>
#include <hip/hip_bf16.h>
#include <math.h>
#include <cstdint>

// ---------------------------------------------------------------------------
// CSWin block for MI455X (gfx1250, wave32, WMMA).
// - All matmuls: v_wmma_f32_16x16x32_f16 (f16 operands, f32 accumulate).
// - GEMM: 128x128 block tile, double-buffered LDS, A staged with
//   global_load_async_to_lds_b128 (ASYNCcnt), 8 WMMAs / wave / K-step.
// - Attention: flash-style online softmax; S/P never touch HBM.
// ---------------------------------------------------------------------------

typedef __attribute__((ext_vector_type(16))) _Float16 v16h;
typedef __attribute__((ext_vector_type(8)))  _Float16 v8h;
typedef __attribute__((ext_vector_type(4)))  _Float16 v4h;
typedef __attribute__((ext_vector_type(8)))  float    v8f;

union AFrag { v16h v; _Float16 h[16]; unsigned u[8]; };
union CFrag { v8f  v; float f[8]; };
union H8    { v8h  v; _Float16 h[8]; };
union H4    { v4h  v; _Float16 h[4]; unsigned u[2]; };

static __device__ __forceinline__ v8f wmma16(v16h a, v16h b, v8f c) {
  // D = A(16x32 f16) * B(32x16 f16) + C(16x16 f32)
  return __builtin_amdgcn_wmma_f32_16x16x32_f16(
      /*neg_a=*/false, a, /*neg_b=*/false, b,
      /*c_mod=*/(short)0, c, /*reuse_a=*/false, /*reuse_b=*/false);
}

// ISA 7.12.2: 16-bit fragment, VGPR vi holds K pair at
// kb = (vi/4)*16 + (lane/16)*8 + (vi%4)*2 for row/col = lane%16.
static __device__ __forceinline__ int frag_kb(int vi, int lh) {
  return (vi >> 2) * 16 + lh * 8 + (vi & 3) * 2;
}

// CDNA5 async global->LDS copy (16B per lane), tracked by ASYNCcnt.
static __device__ __forceinline__ void async_b128(unsigned lds_off, const void* g) {
  unsigned long long ga = (unsigned long long)(uintptr_t)g;
  asm volatile("global_load_async_to_lds_b128 %0, %1, off"
               :: "v"(lds_off), "v"(ga) : "memory");
}
static __device__ __forceinline__ void wait_async0() {
  asm volatile("s_wait_asynccnt 0x0" ::: "memory");
}
// Flat LDS aperture: low 32 bits of a generic pointer to __shared__ are the
// wave-relative LDS byte address (ISA 10.2 aperture translation).
static __device__ __forceinline__ unsigned lds_off(const void* p) {
  return (unsigned)(uintptr_t)p;
}

// -------------------------------- constants --------------------------------
#define BATCH 8
#define HH    56
#define WW    56
#define LTOK  (HH * WW)          // 3136
#define CDIM  256
#define CB    128                // per-branch channels
#define HID   1024
#define MROW  (BATCH * LTOK)     // 25088
#define NWIN  392                // window tokens = 56*7
#define DHEAD 32

// ------------------------------ f32 -> f16 ---------------------------------
__global__ void cvt_kernel(const float* __restrict__ s, _Float16* __restrict__ d, int n) {
  int i = blockIdx.x * blockDim.x + threadIdx.x;
  if (i < n) d[i] = (_Float16)s[i];
}

// ------------------------------- LayerNorm ---------------------------------
__global__ __launch_bounds__(256) void ln_kernel(
    const float* __restrict__ x, const float* __restrict__ g,
    const float* __restrict__ be, _Float16* __restrict__ out) {
  __shared__ float s1[256];
  __shared__ float s2[256];
  const int row = blockIdx.x, t = threadIdx.x;
  float v = x[(size_t)row * CDIM + t];
  s1[t] = v; s2[t] = v * v;
  __syncthreads();
  for (int s = 128; s > 0; s >>= 1) {
    if (t < s) { s1[t] += s1[t + s]; s2[t] += s2[t + s]; }
    __syncthreads();
  }
  float mean = s1[0] * (1.0f / CDIM);
  float var  = s2[0] * (1.0f / CDIM) - mean * mean;
  float r    = rsqrtf(var + 1e-5f);
  out[(size_t)row * CDIM + t] = (_Float16)((v - mean) * r * g[t] + be[t]);
}

// ------------------------- WMMA GEMM (128x128 tile) ------------------------
// C[M,N] = A[M,K](f16) @ B[K,N](f16)  [+bias] [GELU] [+resid(f32)]
// 8 waves: wave (wm in 0..1, wn in 0..3) owns rows wm*64 (4 tiles),
// cols wn*32 (2 tiles) -> 8 WMMAs per K-step. Double-buffered LDS;
// A staged by async-to-LDS, B transposed through registers.
template <bool BIAS, bool GELUF, bool RESID, bool OUTF16>
__global__ __launch_bounds__(256) void gemm_kernel(
    const _Float16* __restrict__ A, const _Float16* __restrict__ Bm,
    const float* __restrict__ bias, const float* __restrict__ resid,
    _Float16* __restrict__ outh, float* __restrict__ outf,
    int M, int N, int K) {
  __shared__ __align__(16) _Float16 lA[2][128 * 32];   // [row][k]
  __shared__ __align__(16) _Float16 lBt[2][128 * 32];  // [n][k] (transposed)
  const int tid  = threadIdx.x;
  const int wave = tid >> 5, lane = tid & 31;
  const int lh   = lane >> 4, lr = lane & 15;
  const int wm   = wave & 1, wn = wave >> 1;
  const int bm   = blockIdx.y * 128, bn = blockIdx.x * 128;

  CFrag acc[4][2];
#pragma unroll
  for (int rt = 0; rt < 4; ++rt)
#pragma unroll
    for (int bt = 0; bt < 2; ++bt)
#pragma unroll
      for (int i = 0; i < 8; ++i) acc[rt][bt].f[i] = 0.0f;

  const int ar  = tid >> 1;          // 0..127 (A row)
  const int ac  = (tid & 1) * 16;    // 0/16   (A k)
  const int bkr = tid >> 3;          // 0..31  (B k row)
  const int bnc = (tid & 7) * 16;    // 0..112 (B n col)

  const _Float16* gA = A + (size_t)(bm + ar) * K + ac;
  const _Float16* gB = Bm + (size_t)bkr * N + bn + bnc;

  H8 br0, br1;
  auto stageA = [&](int buf, int k0) {
    unsigned l0 = lds_off(&lA[buf][ar * 32 + ac]);
    async_b128(l0,      gA + k0);
    async_b128(l0 + 16, gA + k0 + 8);
  };
  auto loadB = [&](int k0) {
    br0.v = *(const v8h*)(gB + (size_t)k0 * N);
    br1.v = *(const v8h*)(gB + (size_t)k0 * N + 8);
  };
  auto storeB = [&](int buf) {
#pragma unroll
    for (int j = 0; j < 8; ++j) {
      lBt[buf][(bnc + j) * 32 + bkr]     = br0.h[j];
      lBt[buf][(bnc + 8 + j) * 32 + bkr] = br1.h[j];
    }
  };

  stageA(0, 0);
  loadB(0);
  storeB(0);
  wait_async0();
  __syncthreads();

  int cur = 0;
  for (int k0 = 0; k0 < K; k0 += 32) {
    const bool more = (k0 + 32 < K);
    if (more) { stageA(cur ^ 1, k0 + 32); loadB(k0 + 32); }

    AFrag af[4], bf[2];
#pragma unroll
    for (int rt = 0; rt < 4; ++rt)
#pragma unroll
      for (int vi = 0; vi < 8; ++vi)
        af[rt].u[vi] =
            *(const unsigned*)&lA[cur][(wm * 64 + rt * 16 + lr) * 32 + frag_kb(vi, lh)];
#pragma unroll
    for (int bt = 0; bt < 2; ++bt)
#pragma unroll
      for (int vi = 0; vi < 8; ++vi)
        bf[bt].u[vi] =
            *(const unsigned*)&lBt[cur][(wn * 32 + bt * 16 + lr) * 32 + frag_kb(vi, lh)];
#pragma unroll
    for (int rt = 0; rt < 4; ++rt)
#pragma unroll
      for (int bt = 0; bt < 2; ++bt)
        acc[rt][bt].v = wmma16(af[rt].v, bf[bt].v, acc[rt][bt].v);

    if (more) { storeB(cur ^ 1); wait_async0(); }
    __syncthreads();
    cur ^= 1;
  }

#pragma unroll
  for (int rt = 0; rt < 4; ++rt)
#pragma unroll
    for (int bt = 0; bt < 2; ++bt)
#pragma unroll
      for (int i = 0; i < 8; ++i) {
        const int gr = bm + wm * 64 + rt * 16 + lh * 8 + i;
        const int gc = bn + wn * 32 + bt * 16 + lr;
        float val = acc[rt][bt].f[i];
        if (BIAS)  val += bias[gc];
        if (GELUF) val = 0.5f * val * (1.0f + erff(val * 0.70710678118654752f));
        if (RESID) val += resid[(size_t)gr * N + gc];
        if (OUTF16) outh[(size_t)gr * N + gc] = (_Float16)val;
        else        outf[(size_t)gr * N + gc] = val;
      }
}

// ---------------------- window token -> flat L index -----------------------
__device__ __forceinline__ int tok2l(int branch, int winB, int t) {
  if (branch == 0) {              // 56x7 stripes
    int y = t / 7, xl = t - y * 7;
    return y * WW + winB * 7 + xl;
  }
  int yl = t / 56, xr = t - yl * 56;  // 7x56 stripes
  return (winB * 7 + yl) * WW + xr;
}

// --------------------------- LePE depthwise conv ---------------------------
__global__ void lepe_kernel(const _Float16* __restrict__ qkv,
                            const float* __restrict__ w0, const float* __restrict__ b0,
                            const float* __restrict__ w1, const float* __restrict__ b1,
                            _Float16* __restrict__ lepe) {
  int gid = blockIdx.x * blockDim.x + threadIdx.x;
  if (gid >= MROW * CDIM) return;
  const int c   = gid & (CDIM - 1);
  const int tok = gid >> 8;
  const int b   = tok / LTOK, l = tok - b * LTOK;
  const int y   = l / WW, x = l - y * WW;
  const int branch = c >> 7, cc = c & (CB - 1);
  const float* w  = branch ? w1 : w0;
  const float* bb = branch ? b1 : b0;
  int yl, xl, ybase, xbase, Hs, Ws;
  if (branch == 0) { Hs = 56; Ws = 7;  yl = y;     xl = x % 7; ybase = 0;      xbase = x - xl; }
  else             { Hs = 7;  Ws = 56; yl = y % 7; xl = x;     ybase = y - yl; xbase = 0; }
  float acc = bb[cc];
#pragma unroll
  for (int dy = -1; dy <= 1; ++dy)
#pragma unroll
    for (int dx = -1; dx <= 1; ++dx) {
      int yy = yl + dy, xx = xl + dx;
      if (yy >= 0 && yy < Hs && xx >= 0 && xx < Ws) {
        int ls = (ybase + yy) * WW + (xbase + xx);
        float vv = (float)qkv[((size_t)(b * LTOK + ls)) * 768 + 512 + c];
        acc += vv * w[((dy + 1) * 3 + (dx + 1)) * CB + cc];
      }
    }
  lepe[(size_t)tok * CDIM + c] = (_Float16)acc;
}

// ----------------------- flash attention (per window) ----------------------
// grid: (qtile 0..3, win*4+head 0..255, branch 0..1); block 256 = 8 waves.
// Each wave: 16 query rows; 13 key steps of 32 (keys padded 392->416).
__global__ __launch_bounds__(256) void attn_kernel(
    const _Float16* __restrict__ qkv, const _Float16* __restrict__ lepe,
    _Float16* __restrict__ outh) {
  __shared__ __align__(16) _Float16 Kt[32 * 32];      // [key][d]
  __shared__ __align__(16) _Float16 Vt[32 * 32];      // [chan][key]
  __shared__ __align__(16) _Float16 Pst[8][16 * 32];  // per-wave [row][key]
  const int tid  = threadIdx.x;
  const int wave = tid >> 5, lane = tid & 31;
  const int lh   = lane >> 4, lr = lane & 15;
  const int branch = blockIdx.z;
  const int wh   = blockIdx.y;
  const int win  = wh >> 2, head = wh & 3;
  const int bIdx = win >> 3, winB = win & 7;
  const int chOff = branch * CB + head * DHEAD;
  const float scale = 0.17677669529663687f;  // 32^-0.5
  const int qbase = blockIdx.x * 128 + wave * 16;

  // Q fragment (A layout), d = 32 = one full WMMA K step.
  AFrag qf;
  {
    int tq = qbase + lr; if (tq > NWIN - 1) tq = NWIN - 1;
    const int lq = tok2l(branch, winB, tq);
    const _Float16* qp = qkv + ((size_t)(bIdx * LTOK + lq)) * 768 + chOff;
#pragma unroll
    for (int vi = 0; vi < 8; ++vi)
      qf.u[vi] = *(const unsigned*)(qp + frag_kb(vi, lh));
  }

  CFrag o0, o1;
  float mrow[8], lsum[8];
#pragma unroll
  for (int i = 0; i < 8; ++i) { o0.f[i] = 0.0f; o1.f[i] = 0.0f; mrow[i] = -1e30f; lsum[i] = 0.0f; }

  for (int ks = 0; ks < 13; ++ks) {
    __syncthreads();
    {  // stage K (row-major) and V (transposed) tiles: 32 keys x 32 d
      const int idx = tid * 4;
      const int key = idx >> 5, dd = idx & 31;
      const int tk  = ks * 32 + key;
      H4 kk, vv;
      if (tk < NWIN) {
        const int lk = tok2l(branch, winB, tk);
        const _Float16* kp = qkv + ((size_t)(bIdx * LTOK + lk)) * 768 + 256 + chOff + dd;
        kk.v = *(const v4h*)kp;
        vv.v = *(const v4h*)(kp + 256);
      } else { kk.u[0] = kk.u[1] = 0u; vv.u[0] = vv.u[1] = 0u; }
      *(v4h*)&Kt[key * 32 + dd] = kk.v;
#pragma unroll
      for (int j = 0; j < 4; ++j) Vt[(dd + j) * 32 + key] = vv.h[j];
    }
    __syncthreads();

    // S = Q * K^T (two 16x16 tiles over this 32-key step)
    AFrag kb0, kb1;
#pragma unroll
    for (int vi = 0; vi < 8; ++vi) {
      const int kb = frag_kb(vi, lh);
      kb0.u[vi] = *(const unsigned*)&Kt[lr * 32 + kb];
      kb1.u[vi] = *(const unsigned*)&Kt[(16 + lr) * 32 + kb];
    }
    CFrag s0, s1, z;
#pragma unroll
    for (int i = 0; i < 8; ++i) z.f[i] = 0.0f;
    s0.v = wmma16(qf.v, kb0.v, z.v);
    s1.v = wmma16(qf.v, kb1.v, z.v);

    const int key0 = ks * 32 + lr, key1 = key0 + 16;
#pragma unroll
    for (int i = 0; i < 8; ++i) {
      s0.f[i] = (key0 < NWIN) ? s0.f[i] * scale : -1e30f;
      s1.f[i] = (key1 < NWIN) ? s1.f[i] * scale : -1e30f;
    }
    // online softmax: row = lh*8+i; cols spread over the 16-lane group
#pragma unroll
    for (int i = 0; i < 8; ++i) {
      float mx = fmaxf(s0.f[i], s1.f[i]);
#pragma unroll
      for (int off = 1; off < 16; off <<= 1) mx = fmaxf(mx, __shfl_xor(mx, off, 32));
      const float mnew = fmaxf(mrow[i], mx);
      const float resc = __expf(mrow[i] - mnew);
      mrow[i] = mnew;
      s0.f[i] = __expf(s0.f[i] - mnew);
      s1.f[i] = __expf(s1.f[i] - mnew);
      float sum = s0.f[i] + s1.f[i];
#pragma unroll
      for (int off = 1; off < 16; off <<= 1) sum += __shfl_xor(sum, off, 32);
      lsum[i] = lsum[i] * resc + sum;
      o0.f[i] *= resc;
      o1.f[i] *= resc;
    }
    // stage P (C layout -> A layout) via per-wave LDS tile
#pragma unroll
    for (int i = 0; i < 8; ++i) {
      const int r = lh * 8 + i;
      Pst[wave][r * 32 + lr]      = (_Float16)s0.f[i];
      Pst[wave][r * 32 + 16 + lr] = (_Float16)s1.f[i];
    }
    __syncthreads();

    // O += P * V (two 16-channel tiles)
    AFrag pa, vb0, vb1;
#pragma unroll
    for (int vi = 0; vi < 8; ++vi) {
      const int kb = frag_kb(vi, lh);
      pa.u[vi]  = *(const unsigned*)&Pst[wave][lr * 32 + kb];
      vb0.u[vi] = *(const unsigned*)&Vt[lr * 32 + kb];
      vb1.u[vi] = *(const unsigned*)&Vt[(16 + lr) * 32 + kb];
    }
    o0.v = wmma16(pa.v, vb0.v, o0.v);
    o1.v = wmma16(pa.v, vb1.v, o1.v);
  }

  // epilogue: normalize, add LePE, scatter to [B,L,C]
#pragma unroll
  for (int i = 0; i < 8; ++i) {
    const int tq = qbase + lh * 8 + i;
    if (tq < NWIN) {
      const float inv = 1.0f / lsum[i];
      const int lq = tok2l(branch, winB, tq);
      const size_t base = ((size_t)(bIdx * LTOK + lq)) * CDIM + chOff;
      outh[base + lr]      = (_Float16)(o0.f[i] * inv + (float)lepe[base + lr]);
      outh[base + 16 + lr] = (_Float16)(o1.f[i] * inv + (float)lepe[base + 16 + lr]);
    }
  }
}

// -------------------------------- launcher ---------------------------------
extern "C" void kernel_launch(void* const* d_in, const int* in_sizes, int n_in,
                              void* d_out, int out_size, void* d_ws, size_t ws_size,
                              hipStream_t stream) {
  (void)in_sizes; (void)n_in; (void)out_size; (void)ws_size;
  const float* x     = (const float*)d_in[0];
  const float* n1g   = (const float*)d_in[3];
  const float* n1b   = (const float*)d_in[4];
  const float* qkvw  = (const float*)d_in[5];
  const float* projw = (const float*)d_in[6];
  const float* projb = (const float*)d_in[7];
  const float* c0w   = (const float*)d_in[8];
  const float* c0b   = (const float*)d_in[9];
  const float* c1w   = (const float*)d_in[10];
  const float* c1b   = (const float*)d_in[11];
  const float* n2g   = (const float*)d_in[12];
  const float* n2b   = (const float*)d_in[13];
  const float* fc1w  = (const float*)d_in[14];
  const float* fc1b  = (const float*)d_in[15];
  const float* fc2w  = (const float*)d_in[16];
  const float* fc2b  = (const float*)d_in[17];
  float* out = (float*)d_out;

  char* ws = (char*)d_ws;
  size_t off = 0;
  auto alloc = [&](size_t bytes) -> char* {
    char* p = ws + off;
    off += (bytes + 255) & ~(size_t)255;
    return p;
  };
  _Float16* qkvw_h  = (_Float16*)alloc((size_t)CDIM * 768 * 2);
  _Float16* projw_h = (_Float16*)alloc((size_t)CDIM * CDIM * 2);
  _Float16* fc1w_h  = (_Float16*)alloc((size_t)CDIM * HID * 2);
  _Float16* fc2w_h  = (_Float16*)alloc((size_t)HID * CDIM * 2);
  _Float16* xn_h    = (_Float16*)alloc((size_t)MROW * CDIM * 2);
  _Float16* qkv_h   = (_Float16*)alloc((size_t)MROW * 768 * 2);
  _Float16* lepe_h  = (_Float16*)alloc((size_t)MROW * CDIM * 2);
  _Float16* attn_h  = (_Float16*)alloc((size_t)MROW * CDIM * 2);
  float*    xa      = (float*)   alloc((size_t)MROW * CDIM * 4);
  _Float16* y_h     = (_Float16*)alloc((size_t)MROW * CDIM * 2);
  _Float16* h1_h    = (_Float16*)alloc((size_t)MROW * HID * 2);

  auto cvt = [&](const float* s, _Float16* d, int n) {
    cvt_kernel<<<(n + 255) / 256, 256, 0, stream>>>(s, d, n);
  };
  cvt(qkvw,  qkvw_h,  CDIM * 768);
  cvt(projw, projw_h, CDIM * CDIM);
  cvt(fc1w,  fc1w_h,  CDIM * HID);
  cvt(fc2w,  fc2w_h,  HID * CDIM);

  // LN1 -> f16
  ln_kernel<<<MROW, 256, 0, stream>>>(x, n1g, n1b, xn_h);
  // QKV = LN(x) @ qkv_w   [25088 x 768]
  gemm_kernel<false, false, false, true>
      <<<dim3(768 / 128, MROW / 128), 256, 0, stream>>>(
          xn_h, qkvw_h, nullptr, nullptr, qkv_h, nullptr, MROW, 768, CDIM);
  // LePE depthwise conv on V
  lepe_kernel<<<(MROW * CDIM + 255) / 256, 256, 0, stream>>>(
      qkv_h, c0w, c0b, c1w, c1b, lepe_h);
  // Flash attention, both branches, O + LePE -> f16 [B,L,C]
  attn_kernel<<<dim3(4, 256, 2), 256, 0, stream>>>(qkv_h, lepe_h, attn_h);
  // x = x + attn @ proj_w + proj_b   (fp32 residual)
  gemm_kernel<true, false, true, false>
      <<<dim3(CDIM / 128, MROW / 128), 256, 0, stream>>>(
          attn_h, projw_h, projb, x, nullptr, xa, MROW, CDIM, CDIM);
  // LN2 -> f16
  ln_kernel<<<MROW, 256, 0, stream>>>(xa, n2g, n2b, y_h);
  // h1 = GELU(y @ fc1_w + fc1_b)
  gemm_kernel<true, true, false, true>
      <<<dim3(HID / 128, MROW / 128), 256, 0, stream>>>(
          y_h, fc1w_h, fc1b, nullptr, h1_h, nullptr, MROW, HID, CDIM);
  // out = xa + h1 @ fc2_w + fc2_b
  gemm_kernel<true, false, true, false>
      <<<dim3(CDIM / 128, MROW / 128), 256, 0, stream>>>(
          h1_h, fc2w_h, fc2b, xa, nullptr, out, MROW, CDIM, HID);
}
